// _MultiHeadAttentionBase_62113817035128
// MI455X (gfx1250) — matrix operand order repaired and checked by reading, not compiled, not run on hardware
//
#include <hip/hip_runtime.h>

#define D_MODEL  1024
#define N_HEADS  16
#define HEAD_DIM 64
#define BATCH    4
#define SEQ      2048

typedef _Float16 half_t;
typedef __attribute__((ext_vector_type(16))) _Float16 v16h;
typedef __attribute__((ext_vector_type(8)))  float    v8f;
typedef __attribute__((ext_vector_type(4)))  int      v4i;

#if defined(__gfx1250__) && __has_builtin(__builtin_amdgcn_global_load_async_to_lds_b128) && __has_builtin(__builtin_amdgcn_s_wait_asynccnt)
#define MHA_HAS_ASYNC_LDS 1
typedef __attribute__((address_space(1))) v4i as1_v4i;
typedef __attribute__((address_space(3))) v4i as3_v4i;
#else
#define MHA_HAS_ASYNC_LDS 0
#endif

#define WMMA_F16(A, B, C) \
    __builtin_amdgcn_wmma_f32_16x16x32_f16(false, (A), false, (B), (short)0, (C), false, false)

// ---------------------------------------------------------------------------
// WMMA fragment loaders (layouts per CDNA5 ISA 7.12.2, wave32):
//  A (16x32 f16): lane<16: row=lane,     elems 0-7 = K 0-7,  8-15 = K 16-23
//                 lane>=16: row=lane-16, elems 0-7 = K 8-15, 8-15 = K 24-31
//    caller passes p = &row[k0 + hi*8]; we load p[0..7] and p[16..23]
//  B (32x16 f16): same K layout as A, N=lane%16
//    caller passes p = &col[k0 + hi*8]; we load p[0..7] and p[16..23]
// ---------------------------------------------------------------------------
__device__ __forceinline__ v16h frag_a_from(const half_t* p) {
    v16h a;
    ((float4*)&a)[0] = *(const float4*)(p);
    ((float4*)&a)[1] = *(const float4*)(p + 16);
    return a;
}
__device__ __forceinline__ v16h frag_b_from(const half_t* p) {
    v16h b;
    ((float4*)&b)[0] = *(const float4*)(p);
    ((float4*)&b)[1] = *(const float4*)(p + 16);
    return b;
}
// A-fragment built from an f32 row (for attn @ V: softmax output is f32)
__device__ __forceinline__ v16h frag_a_from_f32(const float* p) {
    v16h a;
    #pragma unroll
    for (int j = 0; j < 8; ++j) a[j] = (half_t)p[j];
    #pragma unroll
    for (int j = 0; j < 8; ++j) a[j + 8] = (half_t)p[j + 16];
    return a;
}

// ---------------------------------------------------------------------------
// Projection GEMM: Out = alpha * (X[M,K] @ W[K,N] + bias[N])
// 256 threads = 8 waves (4 in M x 2 in N); each wave: 32x32 via 2x2 WMMA frags.
// Block tile: 128(M) x 64(N); K streamed in chunks of 32 through LDS.
// OUT_MODE: 0 = f32 row-major, 1 = f16 row-major, 2 = f16 per-head transposed
//           (VpT[b][h][d][n], so attn@V can stream B-fragments from global)
// ---------------------------------------------------------------------------
template <int IN_F16, int OUT_MODE>
__global__ __launch_bounds__(256)
void mha_proj_kernel(const void* __restrict__ Xv, const float* __restrict__ W,
                     const float* __restrict__ bias, void* __restrict__ Outv,
                     float alpha, int M, int N, int K)
{
    __shared__ half_t As[128][40];  // [m][k], 80B row stride keeps 16B alignment
    __shared__ half_t Bs[64][40];   // [n][k] (W transposed into LDS)

    const float*  Xf = (const float*)Xv;
    const half_t* Xh = (const half_t*)Xv;
    float*  Of = (float*)Outv;
    half_t* Oh = (half_t*)Outv;

    const int tid  = threadIdx.x;
    const int lane = tid & 31;
    const int wid  = tid >> 5;         // 0..7
    const int wm   = wid >> 1;         // 0..3  (M)
    const int wn   = wid & 1;          // 0..1  (N)
    const int r    = lane & 15;
    const int hi   = lane >> 4;
    const int n0   = blockIdx.x * 64;
    const int m0   = blockIdx.y * 128;

    v8f acc[2][2] = {};

    for (int kb = 0; kb < K; kb += 32) {
        // ---- stage A tile: 128 rows x 32 halves = 512 16B-chunks, 2/thread
#if MHA_HAS_ASYNC_LDS
        if (IN_F16) {
            #pragma unroll
            for (int c = 0; c < 2; ++c) {
                int chunk = tid + c * 256;
                int row = chunk >> 2, off = (chunk & 3) * 8;
                __builtin_amdgcn_global_load_async_to_lds_b128(
                    (as1_v4i*)(Xh + (size_t)(m0 + row) * K + kb + off),
                    (as3_v4i*)&As[row][off], 0, 0);
            }
            __builtin_amdgcn_s_wait_asynccnt(0);
        } else
#endif
        {
            #pragma unroll
            for (int c = 0; c < 2; ++c) {
                int chunk = tid + c * 256;
                int row = chunk >> 2, off = (chunk & 3) * 8;
                size_t gi = (size_t)(m0 + row) * K + kb + off;
                half_t* lp = &As[row][off];
                if (IN_F16) {
                    *(float4*)lp = *(const float4*)(Xh + gi);
                } else {
                    #pragma unroll
                    for (int j = 0; j < 8; ++j) lp[j] = (half_t)Xf[gi + j];
                }
            }
        }
        // ---- stage W tile transposed: Bs[n][k] = W[kb+k][n0+n]
        {
            const int kk  = tid >> 3;        // 0..31
            const int nn0 = (tid & 7) * 8;   // 0..56
            #pragma unroll
            for (int j = 0; j < 8; ++j)
                Bs[nn0 + j][kk] = (half_t)W[(size_t)(kb + kk) * N + n0 + nn0 + j];
        }
        __syncthreads();

        v16h a0 = frag_a_from(&As[wm * 32 + r][hi * 8]);
        v16h a1 = frag_a_from(&As[wm * 32 + 16 + r][hi * 8]);
        v16h b0 = frag_b_from(&Bs[wn * 32 + r][hi * 8]);
        v16h b1 = frag_b_from(&Bs[wn * 32 + 16 + r][hi * 8]);
        acc[0][0] = WMMA_F16(a0, b0, acc[0][0]);
        acc[0][1] = WMMA_F16(a0, b1, acc[0][1]);
        acc[1][0] = WMMA_F16(a1, b0, acc[1][0]);
        acc[1][1] = WMMA_F16(a1, b1, acc[1][1]);
        __syncthreads();
    }

    // ---- epilogue
    #pragma unroll
    for (int im = 0; im < 2; ++im) {
        #pragma unroll
        for (int in = 0; in < 2; ++in) {
            const int row0 = m0 + wm * 32 + im * 16 + hi * 8;   // + i
            const int col  = n0 + wn * 32 + in * 16 + r;
            const float bcol = bias[col];
            if (OUT_MODE == 2) {
                // VpT[b][h][d][n]: i walks n -> contiguous 16B store per lane
                const int bidx  = row0 / SEQ;
                const int nbase = row0 - bidx * SEQ;
                const int hh = col >> 6, dd = col & (HEAD_DIM - 1);
                half_t* pt = Oh + ((size_t)(bidx * N_HEADS + hh) * HEAD_DIM + dd) * SEQ + nbase;
                alignas(16) half_t tmp[8];
                #pragma unroll
                for (int i = 0; i < 8; ++i)
                    tmp[i] = (half_t)(alpha * (acc[im][in][i] + bcol));
                *(float4*)pt = *(const float4*)tmp;
            } else if (OUT_MODE == 1) {
                #pragma unroll
                for (int i = 0; i < 8; ++i)
                    Oh[(size_t)(row0 + i) * N + col] =
                        (half_t)(alpha * (acc[im][in][i] + bcol));
            } else {
                #pragma unroll
                for (int i = 0; i < 8; ++i)
                    Of[(size_t)(row0 + i) * N + col] =
                        alpha * (acc[im][in][i] + bcol);
            }
        }
    }
}

// ---------------------------------------------------------------------------
// Scores: S[bh][q][kc] = dot(Qp[bh][q][:], Kp[bh][kc][:])  (K-dim = 64)
// Scale folded into Qp. No LDS: fragments map directly onto row-major f16.
// 128 threads = 4 waves stacked in q; each wave computes a 16x64 strip,
// reusing its A (Q) fragment across 4 B (K) tiles -> 8 WMMAs / wave.
// ---------------------------------------------------------------------------
__global__ __launch_bounds__(128)
void mha_scores_kernel(const half_t* __restrict__ Qp, const half_t* __restrict__ Kp,
                       float* __restrict__ S)
{
    const int tid  = threadIdx.x;
    const int lane = tid & 31;
    const int wave = tid >> 5;
    const int r    = lane & 15;
    const int hi   = lane >> 4;
    const int bh   = blockIdx.z;
    const int b    = bh >> 4;
    const int h    = bh & 15;
    const int q0   = blockIdx.y * 64 + wave * 16;
    const int kc0  = blockIdx.x * 64;

    const half_t* qrow = Qp + (size_t)(b * SEQ + q0 + r) * D_MODEL + h * HEAD_DIM;

    v8f acc[4] = {};
    #pragma unroll
    for (int k0 = 0; k0 < HEAD_DIM; k0 += 32) {
        v16h a = frag_a_from(qrow + k0 + hi * 8);
        #pragma unroll
        for (int t = 0; t < 4; ++t) {
            const half_t* krow =
                Kp + (size_t)(b * SEQ + kc0 + t * 16 + r) * D_MODEL + h * HEAD_DIM;
            v16h bf = frag_b_from(krow + k0 + hi * 8);
            acc[t] = WMMA_F16(a, bf, acc[t]);
        }
    }

    #pragma unroll
    for (int t = 0; t < 4; ++t)
        #pragma unroll
        for (int i = 0; i < 8; ++i)
            S[((size_t)bh * SEQ + q0 + i + hi * 8) * SEQ + kc0 + t * 16 + r] = acc[t][i];
}

// ---------------------------------------------------------------------------
// Row softmax, in place. One 256-thread block per row of SEQ=2048.
// ---------------------------------------------------------------------------
__global__ __launch_bounds__(256)
void mha_softmax_kernel(float* __restrict__ S)
{
    __shared__ float red[256];
    const int t = threadIdx.x;
    float* row = S + (size_t)blockIdx.x * SEQ;

    float vals[8];
    float m = -3.4e38f;
    #pragma unroll
    for (int j = 0; j < 8; ++j) { vals[j] = row[t + j * 256]; m = fmaxf(m, vals[j]); }
    red[t] = m; __syncthreads();
    for (int s = 128; s > 0; s >>= 1) { if (t < s) red[t] = fmaxf(red[t], red[t + s]); __syncthreads(); }
    const float rmax = red[0]; __syncthreads();

    float sum = 0.f;
    #pragma unroll
    for (int j = 0; j < 8; ++j) { vals[j] = __expf(vals[j] - rmax); sum += vals[j]; }
    red[t] = sum; __syncthreads();
    for (int s = 128; s > 0; s >>= 1) { if (t < s) red[t] += red[t + s]; __syncthreads(); }
    const float inv = 1.0f / red[0];

    #pragma unroll
    for (int j = 0; j < 8; ++j) row[t + j * 256] = vals[j] * inv;
}

// ---------------------------------------------------------------------------
// attn @ V: AO[b][q][h*64+d] = sum_k S[bh][q][k] * V[b][k][h*64+d]
// No LDS, no barriers: A-fragments built from the f32 softmax rows with
// inline cvt; B-fragments stream straight from VpT[b][h][d][n] (row-major
// over k). 256 threads = 8 waves (4 in q x 2 in d), 2x2 frags per wave.
// ---------------------------------------------------------------------------
__global__ __launch_bounds__(256)
void mha_attnv_kernel(const float* __restrict__ S, const half_t* __restrict__ VpT,
                      half_t* __restrict__ AO)
{
    const int tid  = threadIdx.x;
    const int lane = tid & 31;
    const int wid  = tid >> 5;     // 0..7
    const int wm   = wid >> 1;     // 0..3 (q)
    const int wn   = wid & 1;      // 0..1 (d)
    const int r    = lane & 15;
    const int hi   = lane >> 4;
    const int bh   = blockIdx.y;
    const int b    = bh >> 4;
    const int h    = bh & 15;
    const int q0   = blockIdx.x * 128 + wm * 32;

    const float*  s0 = S + ((size_t)bh * SEQ + q0 + r) * SEQ;
    const float*  s1 = s0 + (size_t)16 * SEQ;
    const half_t* v0 = VpT + ((size_t)(b * N_HEADS + h) * HEAD_DIM + wn * 32 + r) * SEQ;
    const half_t* v1 = v0 + (size_t)16 * SEQ;

    v8f acc[2][2] = {};
    for (int kb = 0; kb < SEQ; kb += 32) {
        v16h a0 = frag_a_from_f32(s0 + kb + hi * 8);
        v16h a1 = frag_a_from_f32(s1 + kb + hi * 8);
        v16h b0 = frag_b_from(v0 + kb + hi * 8);
        v16h b1 = frag_b_from(v1 + kb + hi * 8);
        acc[0][0] = WMMA_F16(a0, b0, acc[0][0]);
        acc[0][1] = WMMA_F16(a0, b1, acc[0][1]);
        acc[1][0] = WMMA_F16(a1, b0, acc[1][0]);
        acc[1][1] = WMMA_F16(a1, b1, acc[1][1]);
    }

    #pragma unroll
    for (int im = 0; im < 2; ++im)
        #pragma unroll
        for (int in = 0; in < 2; ++in)
            #pragma unroll
            for (int i = 0; i < 8; ++i) {
                int qq = q0 + im * 16 + i + hi * 8;
                int dd = wn * 32 + in * 16 + r;
                AO[(size_t)(b * SEQ + qq) * D_MODEL + h * HEAD_DIM + dd] =
                    (half_t)acc[im][in][i];
            }
}

// ---------------------------------------------------------------------------
extern "C" void kernel_launch(void* const* d_in, const int* in_sizes, int n_in,
                              void* d_out, int out_size, void* d_ws, size_t ws_size,
                              hipStream_t stream)
{
    const float* q  = (const float*)d_in[0];
    const float* k  = (const float*)d_in[1];
    const float* v  = (const float*)d_in[2];
    const float* Wq = (const float*)d_in[3];
    const float* bq = (const float*)d_in[4];
    const float* Wk = (const float*)d_in[5];
    const float* bk = (const float*)d_in[6];
    const float* Wv = (const float*)d_in[7];
    const float* bv = (const float*)d_in[8];
    const float* Wo = (const float*)d_in[9];
    const float* bo = (const float*)d_in[10];

    float* out  = (float*)d_out;
    float* attn = out + (size_t)BATCH * SEQ * D_MODEL;   // attn_weights region

    const int    M      = BATCH * SEQ;                          // 8192
    const size_t f16buf = (size_t)M * D_MODEL * sizeof(half_t); // 16 MB each
    half_t* Qp  = (half_t*)d_ws;
    half_t* Kp  = (half_t*)((char*)d_ws + 1 * f16buf);
    half_t* VpT = (half_t*)((char*)d_ws + 2 * f16buf);  // [b][h][d][n]
    half_t* AO  = (half_t*)((char*)d_ws + 3 * f16buf);

    dim3 pgrid(D_MODEL / 64, M / 128);   // (16, 64)

    // 1/sqrt(HEAD_DIM) folded into the Q projection
    mha_proj_kernel<0, 1><<<pgrid, 256, 0, stream>>>(q, Wq, bq, Qp, 0.125f, M, D_MODEL, D_MODEL);
    mha_proj_kernel<0, 1><<<pgrid, 256, 0, stream>>>(k, Wk, bk, Kp, 1.0f, M, D_MODEL, D_MODEL);
    mha_proj_kernel<0, 2><<<pgrid, 256, 0, stream>>>(v, Wv, bv, VpT, 1.0f, M, D_MODEL, D_MODEL);

    mha_scores_kernel<<<dim3(SEQ / 64, SEQ / 64, BATCH * N_HEADS), 128, 0, stream>>>(Qp, Kp, attn);

    mha_softmax_kernel<<<dim3(BATCH * N_HEADS * SEQ), 256, 0, stream>>>(attn);

    mha_attnv_kernel<<<dim3(SEQ / 128, BATCH * N_HEADS), 256, 0, stream>>>(attn, VpT, AO);

    mha_proj_kernel<1, 0><<<pgrid, 256, 0, stream>>>(AO, Wo, bo, out, 1.0f, M, D_MODEL, D_MODEL);
}
